// MultiAttnBlock_18708877542022
// MI455X (gfx1250) — compile-verified, matrix operand order checked by reading, not run on hardware
//
#include <hip/hip_runtime.h>

typedef __attribute__((ext_vector_type(16))) _Float16 v16h;
typedef __attribute__((ext_vector_type(8)))  _Float16 v8h;
typedef __attribute__((ext_vector_type(8)))  float    v8f;

#define DEV static __device__ __forceinline__

// Problem constants
constexpr int Bb = 8, Tt = 1024, Ee = 128, Hh = 8;
constexpr int Mrows = Bb * Tt;        // 8192
// Workspace offsets (in _Float16 elements); every offset is a multiple of 8 -> 16B aligned
constexpr size_t OFF_XSR   = 0;                         // [8192,128]
constexpr size_t OFF_XSI   = OFF_XSR   + 1048576;
constexpr size_t OFF_WINR  = OFF_XSI   + 1048576;       // [384,128]
constexpr size_t OFF_WINI  = OFF_WINR  + 49152;
constexpr size_t OFF_WOUTR = OFF_WINI  + 49152;         // [128,128]
constexpr size_t OFF_WOUTI = OFF_WOUTR + 16384;
constexpr size_t OFF_WTR   = OFF_WOUTI + 16384;
constexpr size_t OFF_WTI   = OFF_WTR   + 16384;
constexpr size_t OFF_QP    = OFF_WTI   + 16384;         // [B*H,1024,32] packed re|im
constexpr size_t OFF_KP    = OFF_QP    + 2097152;
constexpr size_t OFF_VTR   = OFF_KP    + 2097152;       // [B*H,16,1024] transposed
constexpr size_t OFF_VTI   = OFF_VTR   + 1048576;
constexpr size_t OFF_OR    = OFF_VTI   + 1048576;       // [8192,128]
constexpr size_t OFF_OI    = OFF_OR    + 1048576;
constexpr size_t OFF_O2R   = OFF_OI    + 1048576;
constexpr size_t OFF_O2I   = OFF_O2R   + 1048576;

// Load a 16x32 f16 tile in WMMA A/B operand layout from row-major [rows, ld]
// (A: M x K row-major; B: N x K row-major == B-matrix of K x N).
// Lane m (0-15): row m, K = {kh..kh+7} in v[0:7], {kh+16..kh+23} in v[8:15], kh=0.
// Lane m+16:     row m, kh = 8.
DEV v16h load_tile(const _Float16* __restrict__ base, int row0, int ld, int k0) {
  int lane = threadIdx.x & 31;
  int r = lane & 15;
  int kh = (lane >> 4) * 8;
  const _Float16* p = base + (size_t)(row0 + r) * (size_t)ld + (size_t)(k0 + kh);
  union { v16h v; v8h h[2]; } u;
  u.h[0] = *(const v8h*)(p);
  u.h[1] = *(const v8h*)(p + 16);
  return u.v;
}

DEV v8f wmma_f32(v16h a, v16h b, v8f c) {
  return __builtin_amdgcn_wmma_f32_16x16x32_f16(false, a, false, b, (short)0, c,
                                                false, false);
}

// ---------------- pack / convert kernels ----------------

__global__ void pack_x_kernel(const float* __restrict__ xr, const float* __restrict__ xi,
                              _Float16* __restrict__ xsr, _Float16* __restrict__ xsi) {
  int idx = blockIdx.x * blockDim.x + threadIdx.x;     // over B*E*T
  if (idx >= Bb * Ee * Tt) return;
  int t = idx & (Tt - 1);
  int e = (idx >> 10) & (Ee - 1);
  int b = idx >> 17;
  size_t dst = ((size_t)(b * Tt + t)) * Ee + e;        // [B*T, E]
  xsr[dst] = (_Float16)xr[idx];
  xsi[dst] = (_Float16)xi[idx];
}

__global__ void pack_w_kernel(const float* __restrict__ win_r, const float* __restrict__ win_i,
                              const float* __restrict__ wout_r, const float* __restrict__ wout_i,
                              const float* __restrict__ wt_r, const float* __restrict__ wt_i,
                              _Float16* __restrict__ dwinr, _Float16* __restrict__ dwini,
                              _Float16* __restrict__ dwoutr, _Float16* __restrict__ dwouti,
                              _Float16* __restrict__ dwtr, _Float16* __restrict__ dwti) {
  int i = blockIdx.x * blockDim.x + threadIdx.x;
  if (i < 49152) { dwinr[i] = (_Float16)win_r[i]; dwini[i] = (_Float16)win_i[i]; }
  if (i < 16384) {
    dwoutr[i] = (_Float16)wout_r[i]; dwouti[i] = (_Float16)wout_i[i];
    dwtr[i]   = (_Float16)wt_r[i];   dwti[i]   = (_Float16)wt_i[i];
  }
}

// ---------------- QKV complex GEMM: [8192,128] x [384,128]^T ----------------
// Epilogue scatters into packed-Q/K ([bh,t,32] = re|im, Q pre-scaled by DH^-0.5)
// and transposed V ([bh,d,1024]).

__global__ void gemm_qkv_kernel(const _Float16* __restrict__ xsr, const _Float16* __restrict__ xsi,
                                const _Float16* __restrict__ wr, const _Float16* __restrict__ wi,
                                const float* __restrict__ bin_r, const float* __restrict__ bin_i,
                                _Float16* __restrict__ qp, _Float16* __restrict__ kp,
                                _Float16* __restrict__ vtr, _Float16* __restrict__ vti) {
  int w = threadIdx.x >> 5, lane = threadIdx.x & 31;
  int tile = blockIdx.x * 8 + w;            // 512*24 = 12288 tiles
  int tn = tile % 24, tm = tile / 24;
  int row0 = tm * 16, col0 = tn * 16;
  v8f arr = {}, aii = {}, ari = {}, air = {};
#pragma unroll
  for (int k0 = 0; k0 < 128; k0 += 32) {
    v16h a_r = load_tile(xsr, row0, 128, k0);
    v16h a_i = load_tile(xsi, row0, 128, k0);
    v16h b_r = load_tile(wr, col0, 128, k0);
    v16h b_i = load_tile(wi, col0, 128, k0);
    arr = wmma_f32(a_r, b_r, arr);
    aii = wmma_f32(a_i, b_i, aii);
    ari = wmma_f32(a_r, b_i, ari);
    air = wmma_f32(a_i, b_r, air);
  }
  int col = lane & 15, rowb = (lane >> 4) * 8;
  int j = col0 + col;
  float br = bin_r[j], bi = bin_i[j];
  int mat = col0 >> 7;                      // 0=q, 1=k, 2=v (uniform per wave)
  int h = (col0 >> 4) & 7;                  // head (uniform per wave)
  float qs = (mat == 0) ? 0.25f : 1.0f;     // DH^-0.5 folded into Q
#pragma unroll
  for (int r = 0; r < 8; ++r) {
    int row = row0 + rowb + r;
    int b = row >> 10, t = row & 1023;
    float cre = (arr[r] - aii[r] + br) * qs;
    float cim = (ari[r] + air[r] + bi) * qs;
    if (mat == 0) {
      size_t o = ((size_t)((b * 8 + h) * 1024 + t)) * 32;
      qp[o + col] = (_Float16)cre;  qp[o + 16 + col] = (_Float16)cim;
    } else if (mat == 1) {
      size_t o = ((size_t)((b * 8 + h) * 1024 + t)) * 32;
      kp[o + col] = (_Float16)cre;  kp[o + 16 + col] = (_Float16)cim;
    } else {
      size_t o = ((size_t)((b * 8 + h) * 16 + col)) * 1024 + t;
      vtr[o] = (_Float16)cre;  vti[o] = (_Float16)cim;
    }
  }
}

// ---------------- Flash attention per (b,h) ----------------
// Re(score) = qr.kr + qi.ki == one K=32 WMMA on the packed [re|im] layout.

__global__ void attn_kernel(const _Float16* __restrict__ qp, const _Float16* __restrict__ kp,
                            const _Float16* __restrict__ vtr, const _Float16* __restrict__ vti,
                            _Float16* __restrict__ o_r, _Float16* __restrict__ o_i) {
  __shared__ _Float16 ldsP[8 * 512];        // 1 KB per wave (16x32 f16 P tile)
  int w = threadIdx.x >> 5, lane = threadIdx.x & 31;
  int bh = blockIdx.x >> 3;                 // 64 (b,h) pairs
  int qtile = (blockIdx.x & 7) * 8 + w;     // 64 query tiles per pair
  int t0 = qtile * 16;
  const _Float16* qb = qp  + (size_t)bh * 1024 * 32;
  const _Float16* kb = kp  + (size_t)bh * 1024 * 32;
  const _Float16* vr = vtr + (size_t)bh * 16 * 1024;
  const _Float16* vi = vti + (size_t)bh * 16 * 1024;
  _Float16* myP = ldsP + w * 512;

  v16h aq = load_tile(qb, t0, 32, 0);
  float mrun[8], lrun[8];
  v8f oar = {}, oai = {};
#pragma unroll
  for (int r = 0; r < 8; ++r) { mrun[r] = -1e30f; lrun[r] = 0.0f; }

  int col = lane & 15, rowb = (lane >> 4) * 8;
  v8f zero = {};
  for (int s0 = 0; s0 < 1024; s0 += 32) {   // 32 keys / iteration
    v16h bk0 = load_tile(kb, s0, 32, 0);
    v16h bk1 = load_tile(kb, s0 + 16, 32, 0);
    v8f S0 = wmma_f32(aq, bk0, zero);
    v8f S1 = wmma_f32(aq, bk1, zero);
    float p0[8], p1[8];
#pragma unroll
    for (int r = 0; r < 8; ++r) {
      float mx = fmaxf(S0[r], S1[r]);
#pragma unroll
      for (int msk = 1; msk <= 8; msk <<= 1) mx = fmaxf(mx, __shfl_xor(mx, msk, 32));
      float nm = fmaxf(mrun[r], mx);
      float alpha = __expf(mrun[r] - nm);
      p0[r] = __expf(S0[r] - nm);
      p1[r] = __expf(S1[r] - nm);
      float rs = p0[r] + p1[r];
#pragma unroll
      for (int msk = 1; msk <= 8; msk <<= 1) rs += __shfl_xor(rs, msk, 32);
      lrun[r] = alpha * lrun[r] + rs;
      mrun[r] = nm;
      oar[r] *= alpha;
      oai[r] *= alpha;
    }
    // C-layout -> A-layout transpose through per-wave LDS slice.
#pragma unroll
    for (int r = 0; r < 8; ++r) {
      myP[(rowb + r) * 32 + col]      = (_Float16)p0[r];
      myP[(rowb + r) * 32 + 16 + col] = (_Float16)p1[r];
    }
    __builtin_amdgcn_wave_barrier();        // LDS ops are in-order per wave on CDNA5
    asm volatile("" ::: "memory");
    v16h pA  = load_tile(myP, 0, 32, 0);
    v16h bvr = load_tile(vr, 0, 1024, s0);
    v16h bvi = load_tile(vi, 0, 1024, s0);
    oar = wmma_f32(pA, bvr, oar);
    oai = wmma_f32(pA, bvi, oai);
    __builtin_amdgcn_wave_barrier();
    asm volatile("" ::: "memory");
  }
  int b = bh >> 3, h = bh & 7;
#pragma unroll
  for (int r = 0; r < 8; ++r) {
    int t = t0 + rowb + r;
    float inv = 1.0f / lrun[r];
    size_t o = ((size_t)(b * 1024 + t)) * 128 + h * 16 + col;
    o_r[o] = (_Float16)(oar[r] * inv);
    o_i[o] = (_Float16)(oai[r] * inv);
  }
}

// ---------------- E x E complex GEMM (out-proj / transform) ----------------

template <bool FINAL>
__global__ void gemm_e_kernel(const _Float16* __restrict__ ar, const _Float16* __restrict__ ai,
                              const _Float16* __restrict__ wr, const _Float16* __restrict__ wi,
                              const float* __restrict__ bias_r, const float* __restrict__ bias_i,
                              _Float16* __restrict__ cr, _Float16* __restrict__ ci,
                              float* __restrict__ out) {
  int w = threadIdx.x >> 5, lane = threadIdx.x & 31;
  int tile = blockIdx.x * 8 + w;            // 512*8 = 4096 tiles
  int tn = tile & 7, tm = tile >> 3;
  int row0 = tm * 16, col0 = tn * 16;
  v8f arr = {}, aii = {}, ari = {}, air = {};
#pragma unroll
  for (int k0 = 0; k0 < 128; k0 += 32) {
    v16h a_r = load_tile(ar, row0, 128, k0);
    v16h a_i = load_tile(ai, row0, 128, k0);
    v16h b_r = load_tile(wr, col0, 128, k0);
    v16h b_i = load_tile(wi, col0, 128, k0);
    arr = wmma_f32(a_r, b_r, arr);
    aii = wmma_f32(a_i, b_i, aii);
    ari = wmma_f32(a_r, b_i, ari);
    air = wmma_f32(a_i, b_r, air);
  }
  int col = lane & 15, rowb = (lane >> 4) * 8;
  int j = col0 + col;
  float bre = bias_r[j], bim = bias_i[j];
#pragma unroll
  for (int r = 0; r < 8; ++r) {
    int row = row0 + rowb + r;
    float cre = arr[r] - aii[r] + bre;
    float cim = ari[r] + air[r] + bim;
    if (FINAL) {
      // y[B,T,E] -> out[2,B,C,F,T] with e = c*64+f
      int b = row >> 10, t = row & 1023;
      int c = j >> 6, f = j & 63;
      size_t o = ((size_t)((b * 2 + c) * 64 + f)) * 1024 + t;
      out[o] = cre;
      out[1048576 + o] = cim;
    } else {
      size_t o = (size_t)row * 128 + j;
      cr[o] = (_Float16)cre;
      ci[o] = (_Float16)cim;
    }
  }
}

extern "C" void kernel_launch(void* const* d_in, const int* in_sizes, int n_in,
                              void* d_out, int out_size, void* d_ws, size_t ws_size,
                              hipStream_t stream) {
  const float* x_r    = (const float*)d_in[0];
  const float* x_i    = (const float*)d_in[1];
  const float* win_r  = (const float*)d_in[2];
  const float* win_i  = (const float*)d_in[3];
  const float* bin_r  = (const float*)d_in[4];
  const float* bin_i  = (const float*)d_in[5];
  const float* wout_r = (const float*)d_in[6];
  const float* wout_i = (const float*)d_in[7];
  const float* bout_r = (const float*)d_in[8];
  const float* bout_i = (const float*)d_in[9];
  const float* wt_r   = (const float*)d_in[10];
  const float* wt_i   = (const float*)d_in[11];
  const float* bt_r   = (const float*)d_in[12];
  const float* bt_i   = (const float*)d_in[13];

  _Float16* ws = (_Float16*)d_ws;
  _Float16 *xsr = ws + OFF_XSR,  *xsi = ws + OFF_XSI;
  _Float16 *wnr = ws + OFF_WINR, *wni = ws + OFF_WINI;
  _Float16 *wor = ws + OFF_WOUTR,*woi = ws + OFF_WOUTI;
  _Float16 *wtr = ws + OFF_WTR,  *wti = ws + OFF_WTI;
  _Float16 *qp  = ws + OFF_QP,   *kp  = ws + OFF_KP;
  _Float16 *vtr = ws + OFF_VTR,  *vti = ws + OFF_VTI;
  _Float16 *orr = ws + OFF_OR,   *oii = ws + OFF_OI;
  _Float16 *o2r = ws + OFF_O2R,  *o2i = ws + OFF_O2I;

  pack_x_kernel<<<(Bb * Ee * Tt + 255) / 256, 256, 0, stream>>>(x_r, x_i, xsr, xsi);
  pack_w_kernel<<<(49152 + 255) / 256, 256, 0, stream>>>(
      win_r, win_i, wout_r, wout_i, wt_r, wt_i, wnr, wni, wor, woi, wtr, wti);
  gemm_qkv_kernel<<<1536, 256, 0, stream>>>(xsr, xsi, wnr, wni, bin_r, bin_i,
                                            qp, kp, vtr, vti);
  attn_kernel<<<512, 256, 0, stream>>>(qp, kp, vtr, vti, orr, oii);
  gemm_e_kernel<false><<<512, 256, 0, stream>>>(orr, oii, wor, woi, bout_r, bout_i,
                                                o2r, o2i, nullptr);
  gemm_e_kernel<true><<<512, 256, 0, stream>>>(o2r, o2i, wtr, wti, bt_r, bt_i,
                                               nullptr, nullptr, (float*)d_out);
}
